// NeRFMLP_19301583028501
// MI455X (gfx1250) — matrix operand order repaired and checked by reading, not compiled, not run on hardware
//
#include <hip/hip_runtime.h>

#define NB      262144
#define MTILE   128
#define NTHR    256

typedef _Float16 v16h __attribute__((ext_vector_type(16)));
typedef float    v8f  __attribute__((ext_vector_type(8)));

union H16 { v16h v; uint4 u[2]; };

// ================= Fragment-major activation layout =================
// All LDS activation buffers hold 128 rows. Data is stored as 16B chunks:
//   chunk(kt, ksel, part) at halfword offset (kt*4 + ksel*2 + part)*1024,
//   element (m, off) at + m*8 + off   (off = 0..7 halves)
// where k = kt*32 + part*16 + ksel*8 + off  (matches the 16-bit A-fragment
// layout of CDNA5 ISA §7.12.2). A-fragment ds_load_b128s then have
// consecutive lanes 16B apart -> all 64 LDS banks swept once per half-wave.
__device__ __forceinline__ int fragIdx(int m, int k) {
    const int kt = k >> 5, kk = k & 31;
    return (kt * 4 + ((kk >> 3) & 1) * 2 + ((kk >> 4) & 1)) * 1024 + m * 8 + (kk & 7);
}

// A-fragment: u[0] = chunk(kt,ksel,0)[m], u[1] = chunk(kt,ksel,1)[m]
__device__ __forceinline__ v16h load_afragF(const _Float16* bufKt /* + kt*4096 */,
                                            int m, int ksel) {
    H16 r;
    const _Float16* p = bufKt + ksel * 2048 + m * 8;
    r.u[0] = *(const uint4*)(p);
    r.u[1] = *(const uint4*)(p + 1024);
    return r.v;
}

// B 32x16 f16 fragment: weight row n, 8 halves at k0=ksel*8 and 8 halves at k0+16.
__device__ __forceinline__ v16h load_bfrag(const _Float16* p) {
    H16 r;
    r.u[0] = *(const uint4*)(p);
    r.u[1] = *(const uint4*)(p + 16);
    return r.v;
}

// Writeback: relu -> f16 (packed RTZ convert + packed f16 max), fragment layout.
// nb = per-lane column base (chunk of this lane's output column + off), rows at *8.
__device__ __forceinline__ void store_tileF(v8f acc, _Float16* nb, int r0) {
#pragma unroll
    for (int j = 0; j < 8; j += 2) {
        auto pk = __builtin_amdgcn_cvt_pkrtz(acc[j], acc[j + 1]);
        decltype(pk) z = {};
        pk = __builtin_elementwise_max(pk, z);   // v_pk_max_num_f16 (relu)
        nb[(r0 + j) * 8]     = (_Float16)pk.x;
        nb[(r0 + j + 1) * 8] = (_Float16)pk.y;
    }
}

// One dense layer for the block: out[128][Nout] = relu(in[128][K] @ W^T + b).
// K split across actMain (KT_MAIN k-tiles) and actExtra (KT_EXTRA, skip-concat).
// 8 waves x NT_PER_WAVE N-tiles; two concurrent accumulator chains share B frags.
template <int KT_MAIN, int KT_EXTRA, int NT_PER_WAVE>
__device__ __forceinline__ void mlp_layer(const _Float16* __restrict__ Wh,
                                          const float* __restrict__ bias,
                                          const _Float16* actMain,
                                          const _Float16* actExtra,
                                          _Float16* actOut) {
    constexpr int KT   = KT_MAIN + KT_EXTRA;
    constexpr int KPAD = KT * 32;
    const int lane = threadIdx.x & 31;
    const int wv   = threadIdx.x >> 5;
    const int nlo  = lane & 15;
    const int ksel = lane >> 4;

#pragma unroll 1
    for (int t = 0; t < NT_PER_WAVE; ++t) {
        const int nt = wv + t * 8;
        const int ng = nt * 16 + nlo;
        const float bv = bias[ng];

        // per-lane output-column base in fragment layout
        const int kk = ng & 31;
        _Float16* nb = actOut +
            ((ng >> 5) * 4 + ((kk >> 3) & 1) * 2 + ((kk >> 4) & 1)) * 1024 + (kk & 7);

        v16h bfrag[KT];
#pragma unroll
        for (int kt = 0; kt < KT; ++kt)
            bfrag[kt] = load_bfrag(Wh + (size_t)ng * KPAD + kt * 32 + ksel * 8);

#pragma unroll 1
        for (int ms = 0; ms < 8; ms += 2) {
            v8f acc0, acc1;
#pragma unroll
            for (int j = 0; j < 8; ++j) { acc0[j] = bv; acc1[j] = bv; }

            const int m0 = ms * 16 + nlo;        // chain 0 A-row
            const int m1 = m0 + 16;              // chain 1 A-row
#pragma unroll
            for (int kt = 0; kt < KT_MAIN; ++kt) {
                v16h a0 = load_afragF(actMain + kt * 4096, m0, ksel);
                v16h a1 = load_afragF(actMain + kt * 4096, m1, ksel);
                acc0 = __builtin_amdgcn_wmma_f32_16x16x32_f16(
                           false, a0, false, bfrag[kt], (short)0, acc0, false, false);
                acc1 = __builtin_amdgcn_wmma_f32_16x16x32_f16(
                           false, a1, false, bfrag[kt], (short)0, acc1, false, false);
            }
#pragma unroll
            for (int kt = 0; kt < KT_EXTRA; ++kt) {
                v16h a0 = load_afragF(actExtra + kt * 4096, m0, ksel);
                v16h a1 = load_afragF(actExtra + kt * 4096, m1, ksel);
                acc0 = __builtin_amdgcn_wmma_f32_16x16x32_f16(
                           false, a0, false, bfrag[KT_MAIN + kt], (short)0, acc0, false, false);
                acc1 = __builtin_amdgcn_wmma_f32_16x16x32_f16(
                           false, a1, false, bfrag[KT_MAIN + kt], (short)0, acc1, false, false);
            }
            store_tileF(acc0, nb, ms * 16 + ksel * 8);
            store_tileF(acc1, nb, ms * 16 + 16 + ksel * 8);
        }
    }
    __syncthreads();
}

// ---- weight f32 -> f16 with zero column padding ----
__global__ void convert_pad(const float* __restrict__ src, _Float16* __restrict__ dst,
                            int rows, int scols, int dcols) {
    int i = blockIdx.x * blockDim.x + threadIdx.x;
    int tot = rows * dcols;
    if (i >= tot) return;
    int r = i / dcols, c = i - r * dcols;
    dst[i] = (c < scols) ? (_Float16)src[r * scols + c] : (_Float16)0.f;
}

// ---- fused NeRF forward ----
__global__ __launch_bounds__(NTHR, 4) void nerf_fused(
    const float* __restrict__ x_world, const float* __restrict__ dvec,
    const _Float16* __restrict__ W1h, const _Float16* __restrict__ W2h,
    const _Float16* __restrict__ W3h, const _Float16* __restrict__ W4h,
    const _Float16* __restrict__ W5h, const _Float16* __restrict__ W6h,
    const _Float16* __restrict__ W7h, const _Float16* __restrict__ W8h,
    const _Float16* __restrict__ Wch, const _Float16* __restrict__ Wdh,
    const float* __restrict__ b1, const float* __restrict__ b2,
    const float* __restrict__ b3, const float* __restrict__ b4,
    const float* __restrict__ b5, const float* __restrict__ b6,
    const float* __restrict__ b7, const float* __restrict__ b8,
    const float* __restrict__ Ws, const float* __restrict__ bs,
    const float* __restrict__ bc, const float* __restrict__ bd,
    const float* __restrict__ Wo, const float* __restrict__ bo,
    float* __restrict__ out) {
    extern __shared__ char smem[];
    _Float16* sActA = (_Float16*)smem;                 // 32768 halves (8 kt) = 64KB
    _Float16* sActB = sActA + 32768;                   // 64KB
    _Float16* sPEx  = sActB + 32768;                   // 8192 halves (2 kt) = 16KB
    _Float16* sPEd  = sPEx + 8192;                     // 4096 halves (1 kt) = 8KB
    float*    sWs   = (float*)(sPEd + 4096);           // 256 f32
    float*    sWo   = sWs + 256;                       // 384 f32

    const int tid   = threadIdx.x;
    const int rbase = blockIdx.x * MTILE;

    // preload tiny head weights
    for (int i = tid; i < 256; i += NTHR) sWs[i] = Ws[i];
    for (int i = tid; i < 384; i += NTHR) sWo[i] = Wo[i];

    // ---- positional encodings (threads 0..127, one row each; frag layout) ----
    if (tid < MTILE) {
        const int gr = rbase + tid;
        float fx[3], fd[3];
#pragma unroll
        for (int c = 0; c < 3; ++c) fx[c] = x_world[gr * 3 + c];
#pragma unroll
        for (int c = 0; c < 3; ++c) fd[c] = dvec[gr * 3 + c];
        float nrm = sqrtf(fd[0] * fd[0] + fd[1] * fd[1] + fd[2] * fd[2]) + 1e-8f;
        float inv = 1.f / nrm;
#pragma unroll
        for (int c = 0; c < 3; ++c) fd[c] *= inv;

#pragma unroll
        for (int c = 0; c < 3; ++c) {
            sPEx[fragIdx(tid, c)] = (_Float16)fx[c];
            sPEd[fragIdx(tid, c)] = (_Float16)fd[c];
        }
        float f = 1.f;
        for (int i = 0; i < 10; ++i) {
            int base = 3 + 6 * i;
#pragma unroll
            for (int c = 0; c < 3; ++c) {
                float v = f * fx[c];
                sPEx[fragIdx(tid, base + c)]     = (_Float16)__sinf(v);
                sPEx[fragIdx(tid, base + 3 + c)] = (_Float16)__cosf(v);
            }
            f *= 2.f;
        }
        sPEx[fragIdx(tid, 63)] = (_Float16)0.f;
        f = 1.f;
        for (int i = 0; i < 4; ++i) {
            int base = 3 + 6 * i;
#pragma unroll
            for (int c = 0; c < 3; ++c) {
                float v = f * fd[c];
                sPEd[fragIdx(tid, base + c)]     = (_Float16)__sinf(v);
                sPEd[fragIdx(tid, base + 3 + c)] = (_Float16)__cosf(v);
            }
            f *= 2.f;
        }
        for (int c = 27; c < 32; ++c) sPEd[fragIdx(tid, c)] = (_Float16)0.f;
    }
    __syncthreads();

    // ---- MLP trunk ----
    mlp_layer<2, 0, 2>(W1h, b1, sPEx, nullptr, sActA);   // 63(p64) -> 256
    mlp_layer<8, 0, 2>(W2h, b2, sActA, nullptr, sActB);
    mlp_layer<8, 0, 2>(W3h, b3, sActB, nullptr, sActA);
    mlp_layer<8, 0, 2>(W4h, b4, sActA, nullptr, sActB);
    mlp_layer<8, 2, 2>(W5h, b5, sActB, sPEx, sActA);     // skip concat [h|pe_x]
    mlp_layer<8, 0, 2>(W6h, b6, sActA, nullptr, sActB);
    mlp_layer<8, 0, 2>(W7h, b7, sActB, nullptr, sActA);
    mlp_layer<8, 0, 2>(W8h, b8, sActA, nullptr, sActB);  // h in sActB

    // ---- sigma head: relu(h @ Ws^T + bs), 2 threads per row, chunk-ordered ----
    {
        const int row = tid >> 1, half = tid & 1;
        float s = 0.f;
        for (int c = half * 16; c < half * 16 + 16; ++c) {
            const int k = (c >> 2) * 32 + ((c & 1) << 4) + (((c >> 1) & 1) << 3);
            const _Float16* hp = sActB + c * 1024 + row * 8;
            const float* wp = sWs + k;
#pragma unroll
            for (int o = 0; o < 8; ++o) s += wp[o] * (float)hp[o];
        }
        s += __shfl_xor(s, 1, 32);
        if (half == 0) {
            float sg = s + bs[0];
            out[3 * NB + rbase + row] = fmaxf(sg, 0.f);
        }
    }

    // ---- color branch ----
    mlp_layer<8, 0, 2>(Wch, bc, sActB, nullptr, sActA);  // hc
    mlp_layer<8, 1, 1>(Wdh, bd, sActA, sPEd, sActB);     // [hc|pe_d] -> 128

    // ---- rgb head: sigmoid(hd @ Wo^T + bo), chunk-ordered ----
    if (tid < MTILE) {
        const int gr = rbase + tid;
        float a0 = bo[0], a1 = bo[1], a2 = bo[2];
        for (int c = 0; c < 16; ++c) {
            const int k = (c >> 2) * 32 + ((c & 1) << 4) + (((c >> 1) & 1) << 3);
            const _Float16* hp = sActB + c * 1024 + tid * 8;
#pragma unroll
            for (int o = 0; o < 8; ++o) {
                float h = (float)hp[o];
                a0 += sWo[k + o] * h;
                a1 += sWo[128 + k + o] * h;
                a2 += sWo[256 + k + o] * h;
            }
        }
        out[gr * 3 + 0] = 1.f / (1.f + __expf(-a0));
        out[gr * 3 + 1] = 1.f / (1.f + __expf(-a1));
        out[gr * 3 + 2] = 1.f / (1.f + __expf(-a2));
    }
}

extern "C" void kernel_launch(void* const* d_in, const int* in_sizes, int n_in,
                              void* d_out, int out_size, void* d_ws, size_t ws_size,
                              hipStream_t stream) {
    (void)in_sizes; (void)n_in; (void)out_size; (void)ws_size;
    const float* x_world = (const float*)d_in[0];
    const float* dvec    = (const float*)d_in[1];
    const float* W1 = (const float*)d_in[2];  const float* b1 = (const float*)d_in[3];
    const float* W2 = (const float*)d_in[4];  const float* b2 = (const float*)d_in[5];
    const float* W3 = (const float*)d_in[6];  const float* b3 = (const float*)d_in[7];
    const float* W4 = (const float*)d_in[8];  const float* b4 = (const float*)d_in[9];
    const float* W5 = (const float*)d_in[10]; const float* b5 = (const float*)d_in[11];
    const float* W6 = (const float*)d_in[12]; const float* b6 = (const float*)d_in[13];
    const float* W7 = (const float*)d_in[14]; const float* b7 = (const float*)d_in[15];
    const float* W8 = (const float*)d_in[16]; const float* b8 = (const float*)d_in[17];
    const float* Ws = (const float*)d_in[18]; const float* bs = (const float*)d_in[19];
    const float* Wc = (const float*)d_in[20]; const float* bc = (const float*)d_in[21];
    const float* Wd = (const float*)d_in[22]; const float* bd = (const float*)d_in[23];
    const float* Wo = (const float*)d_in[24]; const float* bo = (const float*)d_in[25];
    float* out = (float*)d_out;

    _Float16* p   = (_Float16*)d_ws;
    _Float16* W1h = p;                 // 256*64
    _Float16* W2h = W1h + 256 * 64;    // 256*256
    _Float16* W3h = W2h + 256 * 256;
    _Float16* W4h = W3h + 256 * 256;
    _Float16* W5h = W4h + 256 * 256;   // 256*320
    _Float16* W6h = W5h + 256 * 320;
    _Float16* W7h = W6h + 256 * 256;
    _Float16* W8h = W7h + 256 * 256;
    _Float16* Wch = W8h + 256 * 256;
    _Float16* Wdh = Wch + 256 * 256;   // 128*288

    auto cvt = [&](const float* s, _Float16* dst, int r, int sc, int dc) {
        int tot = r * dc;
        hipLaunchKernelGGL(convert_pad, dim3((tot + 255) / 256), dim3(256), 0, stream,
                           s, dst, r, sc, dc);
    };
    cvt(W1, W1h, 256, 63, 64);
    cvt(W2, W2h, 256, 256, 256);
    cvt(W3, W3h, 256, 256, 256);
    cvt(W4, W4h, 256, 256, 256);
    cvt(W5, W5h, 256, 319, 320);
    cvt(W6, W6h, 256, 256, 256);
    cvt(W7, W7h, 256, 256, 256);
    cvt(W8, W8h, 256, 256, 256);
    cvt(Wc, Wch, 256, 256, 256);
    cvt(Wd, Wdh, 128, 283, 288);

    const size_t smem = (size_t)(32768 + 32768 + 8192 + 4096) * sizeof(_Float16)
                      + (256 + 384) * sizeof(float);
    hipFuncSetAttribute(reinterpret_cast<const void*>(nerf_fused),
                        hipFuncAttributeMaxDynamicSharedMemorySize, (int)smem);

    hipLaunchKernelGGL(nerf_fused, dim3(NB / MTILE), dim3(NTHR), smem, stream,
                       x_world, dvec,
                       W1h, W2h, W3h, W4h, W5h, W6h, W7h, W8h, Wch, Wdh,
                       b1, b2, b3, b4, b5, b6, b7, b8,
                       Ws, bs, bc, bd, Wo, bo, out);
}
